// RoPEMultiHeadAttention_19275813224606
// MI455X (gfx1250) — matrix operand order repaired and checked by reading, not compiled, not run on hardware
//
#include <hip/hip_runtime.h>
#include <hip/hip_bf16.h>
#include <math.h>

typedef _Float16 h16;
typedef __attribute__((ext_vector_type(16))) _Float16 v16h;
typedef __attribute__((ext_vector_type(8)))  _Float16 v8h;
typedef __attribute__((ext_vector_type(8)))  float    v8f;

#define SEQ   4096
#define DM    512
#define NH    8
#define HD    64
#define ROWS  8192          // B*SEQ

// D = A(16x32 f16) * B(32x16 f16) + C(16x16 f32)
__device__ __forceinline__ v8f wmma16(v16h a, v16h b, v8f c) {
  return __builtin_amdgcn_wmma_f32_16x16x32_f16(false, a, false, b, (short)0, c,
                                                false, false);
}

// A-operand tile (16x32) from a row-major matrix.
// lane = m + 16*half : row m, K = half*16..half*16+15 (contiguous)
__device__ __forceinline__ v16h ld_a_rm(const h16* row, int k0, int half) {
  v8h c0 = *(const v8h*)(row + k0 + half * 16);
  v8h c1 = *(const v8h*)(row + k0 + half * 16 + 8);
  return __builtin_shufflevector(c0, c1, 0,1,2,3,4,5,6,7,8,9,10,11,12,13,14,15);
}

// Async global -> LDS copy, 16B per lane, tracked by ASYNCcnt.
__device__ __forceinline__ void async_b128(void* lds_ptr, const void* gptr) {
  uint32_t loff = (uint32_t)(uintptr_t)lds_ptr;         // addr[31:0] = LDS offset
  asm volatile("global_load_async_to_lds_b128 %0, %1, off"
               :: "v"(loff), "v"((uint64_t)(uintptr_t)gptr)
               : "memory");
}
#define ASYNC_WAIT(n) asm volatile("s_wait_asynccnt %0" :: "i"(n) : "memory")

// ---------------------------------------------------------------- converts
__global__ void cast_f32_f16_kernel(const float* __restrict__ src,
                                    h16* __restrict__ dst, int n) {
  int i = blockIdx.x * blockDim.x + threadIdx.x;
  if (i < n) dst[i] = (h16)src[i];
}

// Wall[1536][512] = rows of Wq | Wk | Wv  (f16)
__global__ void pack_wall_kernel(const float* __restrict__ Wq,
                                 const float* __restrict__ Wk,
                                 const float* __restrict__ Wv,
                                 h16* __restrict__ Wall) {
  int i = blockIdx.x * blockDim.x + threadIdx.x;
  if (i >= 1536 * 512) return;
  int row = i >> 9, col = i & 511;
  int which = row >> 9, rl = row & 511;
  const float* W = (which == 0) ? Wq : (which == 1) ? Wk : Wv;
  Wall[i] = (h16)W[rl * 512 + col];
}

// rope[s][d] = {cos, sin} of s * 10000^(-2d/64), d in [0,32)
__global__ void rope_kernel(float* __restrict__ rope) {
  int i = blockIdx.x * blockDim.x + threadIdx.x;
  if (i >= SEQ * 32) return;
  int s = i >> 5, d = i & 31;
  float freq = powf(10000.0f, -(float)(2 * d) * (1.0f / 64.0f));
  float a = (float)s * freq;
  rope[2 * i + 0] = cosf(a);
  rope[2 * i + 1] = sinf(a);
}

// ------------------------------------------------- QKV GEMM + bias + RoPE
// Block 128 thr = 4 waves; wave does 16x64 of Y[8192][1536] = Xh * Wall^T.
// N-tile (64) is exactly one head of one of q/k/v -> RoPE fused in epilogue.
__global__ __launch_bounds__(128) void qkv_rope_kernel(
    const h16* __restrict__ Xh, const h16* __restrict__ Wall,
    const float* __restrict__ bq, const float* __restrict__ bk,
    const float* __restrict__ bv, const float* __restrict__ rope,
    h16* __restrict__ Qh, h16* __restrict__ Kh, h16* __restrict__ Vt) {
  int wave = threadIdx.x >> 5;
  int tid  = threadIdx.x & 31;
  int nl = tid & 15, kh = tid >> 4;
  int M0 = blockIdx.x * 64 + wave * 16;
  int N0 = blockIdx.y * 64;

  const h16* arow = Xh + (size_t)(M0 + nl) * DM;
  v8f acc[4] = {};
  for (int k0 = 0; k0 < DM; k0 += 32) {
    v16h a = ld_a_rm(arow, k0, kh);
#pragma unroll
    for (int t = 0; t < 4; ++t) {
      const h16* brow = Wall + (size_t)(N0 + t * 16 + nl) * DM;
      v16h b = *(const v16h*)(brow + k0 + kh * 16);   // B column = W row
      acc[t] = wmma16(a, b, acc[t]);
    }
  }

  int which = N0 >> 9;              // 0=q 1=k 2=v
  int nbase = N0 & 511;
  int head  = nbase >> 6;
  const float* bias = (which == 0) ? bq : (which == 1) ? bk : bv;

#pragma unroll
  for (int r = 0; r < 8; ++r) {
    int row = M0 + r + 8 * kh;      // global row = b*SEQ + s
    int b_  = row >> 12;
    int s_  = row & (SEQ - 1);
    if (which < 2) {
      h16* dst = ((which == 0) ? Qh : Kh) +
                 ((size_t)(b_ * NH + head) * SEQ + s_) * HD;
#pragma unroll
      for (int t = 0; t < 2; ++t) {
        int d = t * 16 + nl;        // d < 32, pair (d, d+32)
        float lo = acc[t][r]     + bias[nbase + d];
        float hi = acc[t + 2][r] + bias[nbase + d + 32];
        float c  = rope[((size_t)s_ * 32 + d) * 2 + 0];
        float sn = rope[((size_t)s_ * 32 + d) * 2 + 1];
        dst[d]      = (h16)(lo * c - hi * sn);
        dst[d + 32] = (h16)(hi * c + lo * sn);
      }
    } else {                        // V stored transposed: [bh][d][s]
      h16* dst = Vt + (size_t)(b_ * NH + head) * HD * SEQ + s_;
#pragma unroll
      for (int t = 0; t < 4; ++t) {
        int d = t * 16 + nl;
        dst[(size_t)d * SEQ] = (h16)(acc[t][r] + bias[nbase + d]);
      }
    }
  }
}

// ----------------------------------------------- fused flash attention
// grid (SEQ/64, B*NH); 4 waves/block, each wave owns 16 query rows.
// K/V tiles (32 keys) are double-buffered in LDS via async global->LDS
// copies (ASYNCcnt), shared by all 4 waves: 4x less K/V global traffic and
// copy latency overlapped with softmax/WMMA of the current tile.
__global__ __launch_bounds__(128) void flash_attn_kernel(
    const h16* __restrict__ Qh, const h16* __restrict__ Kh,
    const h16* __restrict__ Vt, h16* __restrict__ Attn) {
  __shared__ alignas(32) _Float16 lds_k[2][32 * HD];   // [j][d]  4KB each
  __shared__ alignas(32) _Float16 lds_v[2][HD * 32];   // [d][j]  4KB each
  __shared__ alignas(32) _Float16 lds_p[4][16 * 32];   // per-wave P transpose

  int wave = threadIdx.x >> 5;
  int tid  = threadIdx.x & 31;
  int nl = tid & 15, hi = tid >> 4;
  int i0b = blockIdx.x * 64;
  int i0  = i0b + wave * 16;
  int bh  = blockIdx.y;

  const h16* Qb = Qh + (size_t)bh * SEQ * HD;
  const h16* Kb = Kh + (size_t)bh * SEQ * HD;
  const h16* Vb = Vt + (size_t)bh * HD * SEQ;

  const h16* qrow = Qb + (size_t)(i0 + nl) * HD;
  v16h Aq0 = ld_a_rm(qrow, 0, hi);
  v16h Aq1 = ld_a_rm(qrow, 32, hi);

  v8f O[4] = {};
  float m_i[8], l_i[8];
#pragma unroll
  for (int r = 0; r < 8; ++r) { m_i[r] = -1e30f; l_i[r] = 0.0f; }

  int nIter = (i0b + 63) / 32 + 1;
  int th = threadIdx.x;

  // issue async copies of K/V tile at j0 into buffer nb
  auto issue_tile = [&](int j0, int nb) {
#pragma unroll
    for (int q = 0; q < 2; ++q) {              // K: 32x64 halfs, contiguous
      int c = q * 128 + th;                    // 16B chunk index 0..255
      async_b128(&lds_k[nb][c * 8], Kb + (size_t)j0 * HD + c * 8);
    }
#pragma unroll
    for (int q = 0; q < 2; ++q) {              // V: 64 rows x 32 j
      int c = q * 128 + th;
      int row = c >> 2, sub = c & 3;
      async_b128(&lds_v[nb][row * 32 + sub * 8],
                 Vb + (size_t)row * SEQ + j0 + sub * 8);
    }
  };

  issue_tile(0, 0);
  for (int it = 0; it < nIter; ++it) {
    int j0  = it * 32;
    int cur = it & 1;
    if (it + 1 < nIter) {
      issue_tile(j0 + 32, cur ^ 1);
      ASYNC_WAIT(4);                 // the 4 older (current-tile) copies done
    } else {
      ASYNC_WAIT(0);
    }
    __syncthreads();                 // all waves' copies of tile `cur` visible

    if (j0 <= i0 + 15) {             // wave-uniform causal skip
      // S = Q * K^T  (B-tile rows from LDS, contiguous per lane)
      v8f S[2] = {};
#pragma unroll
      for (int t = 0; t < 2; ++t) {
        const _Float16* krow = &lds_k[cur][(t * 16 + nl) * HD];
        v16h b0 = *(const v16h*)(krow + hi * 16);
        v16h b1 = *(const v16h*)(krow + 32 + hi * 16);
        S[t] = wmma16(Aq0, b0, S[t]);
        S[t] = wmma16(Aq1, b1, S[t]);
      }

      // online softmax (rows live across the 16 lanes of each half-wave)
      float p0v[8], p1v[8], alpha[8];
#pragma unroll
      for (int r = 0; r < 8; ++r) {
        int i = i0 + r + 8 * hi;
        float s0 = S[0][r] * 0.125f;
        float s1 = S[1][r] * 0.125f;
        if (j0 + nl      > i) s0 = -1e30f;
        if (j0 + 16 + nl > i) s1 = -1e30f;
        float mx = fmaxf(s0, s1);
        mx = fmaxf(mx, __shfl_xor(mx, 1));
        mx = fmaxf(mx, __shfl_xor(mx, 2));
        mx = fmaxf(mx, __shfl_xor(mx, 4));
        mx = fmaxf(mx, __shfl_xor(mx, 8));
        float mn = fmaxf(m_i[r], mx);
        float al = __expf(m_i[r] - mn);
        m_i[r] = mn; alpha[r] = al;
        float p0 = __expf(s0 - mn);
        float p1 = __expf(s1 - mn);
        float rs = p0 + p1;
        rs += __shfl_xor(rs, 1);
        rs += __shfl_xor(rs, 2);
        rs += __shfl_xor(rs, 4);
        rs += __shfl_xor(rs, 8);
        l_i[r] = l_i[r] * al + rs;
        p0v[r] = p0; p1v[r] = p1;
      }

#pragma unroll
      for (int td = 0; td < 4; ++td)
#pragma unroll
        for (int r = 0; r < 8; ++r) O[td][r] *= alpha[r];

      // P: C-layout -> A-layout through per-wave LDS (same-wave DS ops are
      // in-order; only a compiler barrier is needed)
#pragma unroll
      for (int r = 0; r < 8; ++r) {
        lds_p[wave][(r + 8 * hi) * 32 + nl]      = (h16)p0v[r];
        lds_p[wave][(r + 8 * hi) * 32 + 16 + nl] = (h16)p1v[r];
      }
      asm volatile("" ::: "memory");
      const _Float16* prow = &lds_p[wave][nl * 32];
      v8h c0 = *(const v8h*)(prow + hi * 16);
      v8h c1 = *(const v8h*)(prow + hi * 16 + 8);
      v16h Ap = __builtin_shufflevector(c0, c1,
                  0,1,2,3,4,5,6,7,8,9,10,11,12,13,14,15);

      // O += P * V  (B-tile: lane d holds LDS V row d, contiguous)
#pragma unroll
      for (int td = 0; td < 4; ++td) {
        const _Float16* vrow = &lds_v[cur][(td * 16 + nl) * 32];
        v16h bvt = *(const v16h*)(vrow + hi * 16);
        O[td] = wmma16(Ap, bvt, O[td]);
      }
    }
    __syncthreads();                 // tile `cur` free for reuse next iter
  }

  int b_ = bh >> 3, h_ = bh & 7;
#pragma unroll
  for (int r = 0; r < 8; ++r) {
    float inv = 1.0f / l_i[r];
    int i = i0 + r + 8 * hi;
    size_t row = (size_t)(b_ * SEQ + i) * DM + h_ * HD;
#pragma unroll
    for (int td = 0; td < 4; ++td)
      Attn[row + td * 16 + nl] = (h16)(O[td][r] * inv);
  }
}

// ------------------------------------------------- output projection GEMM
__global__ __launch_bounds__(128) void out_proj_kernel(
    const h16* __restrict__ Attn, const h16* __restrict__ Woh,
    const float* __restrict__ bo, float* __restrict__ out) {
  int wave = threadIdx.x >> 5;
  int tid  = threadIdx.x & 31;
  int nl = tid & 15, kh = tid >> 4;
  int M0 = blockIdx.x * 64 + wave * 16;
  int N0 = blockIdx.y * 64;

  const h16* arow = Attn + (size_t)(M0 + nl) * DM;
  v8f acc[4] = {};
  for (int k0 = 0; k0 < DM; k0 += 32) {
    v16h a = ld_a_rm(arow, k0, kh);
#pragma unroll
    for (int t = 0; t < 4; ++t) {
      const h16* brow = Woh + (size_t)(N0 + t * 16 + nl) * DM;
      v16h b = *(const v16h*)(brow + k0 + kh * 16);
      acc[t] = wmma16(a, b, acc[t]);
    }
  }
#pragma unroll
  for (int r = 0; r < 8; ++r) {
    size_t row = (size_t)(M0 + r + 8 * kh) * DM;
#pragma unroll
    for (int t = 0; t < 4; ++t) {
      int n = N0 + t * 16 + nl;
      out[row + n] = acc[t][r] + bo[n];
    }
  }
}

// ----------------------------------------------------------------- launch
extern "C" void kernel_launch(void* const* d_in, const int* in_sizes, int n_in,
                              void* d_out, int out_size, void* d_ws, size_t ws_size,
                              hipStream_t stream) {
  (void)in_sizes; (void)n_in; (void)out_size; (void)ws_size;
  const float* x  = (const float*)d_in[0];
  // d_in[1] = mask (causality recomputed analytically in-kernel)
  const float* Wq = (const float*)d_in[2];
  const float* bq = (const float*)d_in[3];
  const float* Wk = (const float*)d_in[4];
  const float* bk = (const float*)d_in[5];
  const float* Wv = (const float*)d_in[6];
  const float* bv = (const float*)d_in[7];
  const float* Wo = (const float*)d_in[8];
  const float* bo = (const float*)d_in[9];
  float* out = (float*)d_out;

  char* ws = (char*)d_ws;
  size_t off = 0;
  h16*   Xh   = (h16*)(ws + off);   off += (size_t)ROWS * DM * 2;       //  8 MB
  h16*   Wall = (h16*)(ws + off);   off += (size_t)1536 * DM * 2;       // 1.5 MB
  h16*   Woh  = (h16*)(ws + off);   off += (size_t)DM * DM * 2;         // 0.5 MB
  float* rope = (float*)(ws + off); off += (size_t)SEQ * 32 * 2 * 4;    //  1 MB
  h16*   Qh   = (h16*)(ws + off);   off += (size_t)16 * SEQ * HD * 2;   //  8 MB
  h16*   Kh   = (h16*)(ws + off);   off += (size_t)16 * SEQ * HD * 2;   //  8 MB
  h16*   Vt   = (h16*)(ws + off);   off += (size_t)16 * SEQ * HD * 2;   //  8 MB
  h16*   Attn = (h16*)(ws + off);   off += (size_t)ROWS * DM * 2;       //  8 MB

  cast_f32_f16_kernel<<<(ROWS * DM + 255) / 256, 256, 0, stream>>>(x, Xh, ROWS * DM);
  pack_wall_kernel<<<(1536 * DM + 255) / 256, 256, 0, stream>>>(Wq, Wk, Wv, Wall);
  cast_f32_f16_kernel<<<(DM * DM + 255) / 256, 256, 0, stream>>>(Wo, Woh, DM * DM);
  rope_kernel<<<(SEQ * 32 + 255) / 256, 256, 0, stream>>>(rope);

  qkv_rope_kernel<<<dim3(ROWS / 64, 1536 / 64), 128, 0, stream>>>(
      Xh, Wall, bq, bk, bv, rope, Qh, Kh, Vt);

  flash_attn_kernel<<<dim3(SEQ / 64, 16), 128, 0, stream>>>(Qh, Kh, Vt, Attn);

  out_proj_kernel<<<dim3(ROWS / 64, DM / 64), 128, 0, stream>>>(Attn, Woh, bo, out);
}
